// BitSelfAttention_31224412242442
// MI455X (gfx1250) — 16-bit and 8-bit matrix operand order repaired and checked by reading, compiled, not run on hardware
//
#include <hip/hip_runtime.h>


typedef __attribute__((ext_vector_type(16))) _Float16 v16h;
typedef __attribute__((ext_vector_type(8)))  float    v8f;
typedef __attribute__((ext_vector_type(8)))  int      v8i;

union AF { v16h h; v8i i; };

constexpr int    Bn  = 8;
constexpr int    Tn  = 1024;
constexpr int    Dn  = 2048;
constexpr int    Hn  = 16;
constexpr int    HDn = 128;
constexpr int    BTn = Bn * Tn;
constexpr size_t DDsz  = (size_t)Dn * Dn;
constexpr size_t BTDsz = (size_t)BTn * Dn;

constexpr size_t OFF_SCAL = 0;
constexpr size_t OFF_LN   = 8192;
constexpr size_t OFF_XI8  = OFF_LN  + BTDsz * 4;
constexpr size_t OFF_W8   = OFF_XI8 + BTDsz;
constexpr size_t OFF_QH   = OFF_W8  + DDsz * 4;
constexpr size_t OFF_KH   = OFF_QH  + BTDsz * 2;
constexpr size_t OFF_VT   = OFF_KH  + BTDsz * 2;


__global__ void init_kernel(unsigned* sc) {
  if (threadIdx.x < 2) sc[threadIdx.x] = 0u;
}

__global__ __launch_bounds__(256) void ln_kernel(const float* __restrict__ x,
                                                 const float* __restrict__ g,
                                                 const float* __restrict__ be,
                                                 float* __restrict__ xln,
                                                 unsigned* amax) {
  __shared__ float sd[256];
  const int row = blockIdx.x, tid = threadIdx.x;
  const float* xr = x + (size_t)row * Dn;
  float v[8];
  float s = 0.f;
#pragma unroll
  for (int i = 0; i < 8; ++i) { v[i] = xr[tid + i * 256]; s += v[i]; }
  sd[tid] = s; __syncthreads();
  for (int o = 128; o > 0; o >>= 1) { if (tid < o) sd[tid] += sd[tid + o]; __syncthreads(); }
  const float mu = sd[0] * (1.f / Dn);
  __syncthreads();
  float s2 = 0.f;
#pragma unroll
  for (int i = 0; i < 8; ++i) { float d = v[i] - mu; s2 += d * d; }
  sd[tid] = s2; __syncthreads();
  for (int o = 128; o > 0; o >>= 1) { if (tid < o) sd[tid] += sd[tid + o]; __syncthreads(); }
  const float rstd = rsqrtf(sd[0] * (1.f / Dn) + 1e-5f);
  __syncthreads();
  float mx = 0.f;
  float* outr = xln + (size_t)row * Dn;
#pragma unroll
  for (int i = 0; i < 8; ++i) {
    int c = tid + i * 256;
    float y = (v[i] - mu) * rstd * g[c] + be[c];
    outr[c] = y;
    mx = fmaxf(mx, fabsf(y));
  }
  sd[tid] = mx; __syncthreads();
  for (int o = 128; o > 0; o >>= 1) { if (tid < o) sd[tid] = fmaxf(sd[tid], sd[tid + o]); __syncthreads(); }
  if (tid == 0) atomicMax(amax, __float_as_uint(sd[0]));
}

__global__ __launch_bounds__(256) void wabs_kernel(const float* __restrict__ W0,
                                                   const float* __restrict__ W1,
                                                   const float* __restrict__ W2,
                                                   const float* __restrict__ W3,
                                                   float* __restrict__ partials) {
  const float* W = (blockIdx.y == 0) ? W0 : (blockIdx.y == 1) ? W1 : (blockIdx.y == 2) ? W2 : W3;
  __shared__ float sd[256];
  float s = 0.f;
  for (size_t i = (size_t)blockIdx.x * 256 + threadIdx.x; i < DDsz; i += 256u * 256u)
    s += fabsf(W[i]);
  sd[threadIdx.x] = s; __syncthreads();
  for (int o = 128; o > 0; o >>= 1) { if ((int)threadIdx.x < o) sd[threadIdx.x] += sd[threadIdx.x + o]; __syncthreads(); }
  if (threadIdx.x == 0) partials[blockIdx.y * 256 + blockIdx.x] = sd[0];
}

__global__ __launch_bounds__(256) void scales1_kernel(float* sc, const float* partials) {
  __shared__ float sd[256];
  const int tid = threadIdx.x;
  for (int m = 0; m < 4; ++m) {
    sd[tid] = partials[m * 256 + tid]; __syncthreads();
    for (int o = 128; o > 0; o >>= 1) { if (tid < o) sd[tid] += sd[tid + o]; __syncthreads(); }
    if (tid == 0) sc[4 + m] = 1.f / fmaxf(sd[0] * (1.f / (float)DDsz), 1e-5f);
    __syncthreads();
  }
  if (tid == 0) {
    float amax = __uint_as_float(((unsigned*)sc)[0]);
    float xs = 127.f / fmaxf(amax, 1e-5f);
    sc[8] = xs;
    for (int m = 0; m < 3; ++m) sc[12 + m] = 1.f / (xs * sc[4 + m]);
  }
}

__global__ void scales2_kernel(float* sc) {
  float amax = __uint_as_float(((unsigned*)sc)[1]);
  float xs = 127.f / fmaxf(amax, 1e-5f);
  sc[16] = xs;
  sc[15] = 1.f / (xs * sc[7]);
}

__global__ __launch_bounds__(256) void quantw_kernel(const float* __restrict__ W0,
                                                     const float* __restrict__ W1,
                                                     const float* __restrict__ W2,
                                                     const float* __restrict__ W3,
                                                     const float* __restrict__ sc,
                                                     signed char* __restrict__ w8) {
  const int m = blockIdx.y;
  const float* W = (m == 0) ? W0 : (m == 1) ? W1 : (m == 2) ? W2 : W3;
  const float wsc = sc[4 + m];
  signed char* o = w8 + (size_t)m * DDsz;
  for (size_t i = (size_t)blockIdx.x * 256 + threadIdx.x; i < DDsz; i += (size_t)gridDim.x * 256)
    o[i] = (signed char)fminf(1.f, fmaxf(-1.f, rintf(W[i] * wsc)));
}

__global__ __launch_bounds__(256) void quanta_kernel(const float* __restrict__ src,
                                                     signed char* __restrict__ dst,
                                                     const float* __restrict__ scptr,
                                                     size_t n) {
  const float xs = scptr[0];
  for (size_t i = (size_t)blockIdx.x * 256 + threadIdx.x; i < n; i += (size_t)gridDim.x * 256)
    dst[i] = (signed char)fminf(127.f, fmaxf(-127.f, rintf(src[i] * xs)));
}

__global__ __launch_bounds__(256) void amax_kernel(const float* __restrict__ src,
                                                   unsigned* amax, size_t n) {
  __shared__ float sd[256];
  float mx = 0.f;
  for (size_t i = (size_t)blockIdx.x * 256 + threadIdx.x; i < n; i += (size_t)gridDim.x * 256)
    mx = fmaxf(mx, fabsf(src[i]));
  sd[threadIdx.x] = mx; __syncthreads();
  for (int o = 128; o > 0; o >>= 1) { if ((int)threadIdx.x < o) sd[threadIdx.x] = fmaxf(sd[threadIdx.x], sd[threadIdx.x + o]); __syncthreads(); }
  if (threadIdx.x == 0) atomicMax(amax, __float_as_uint(sd[0]));
}

template <int MODE>
__global__ __launch_bounds__(128) void gemm_i8_kernel(const signed char* __restrict__ Xq,
                                                      const signed char* __restrict__ W8,
                                                      const float* __restrict__ bias,
                                                      const float* __restrict__ deqp,
                                                      float* __restrict__ of32,
                                                      _Float16* __restrict__ oh) {
  const int lane = threadIdx.x & 31;
  const int wid  = blockIdx.x * 4 + (threadIdx.x >> 5);
  const int mbase = (wid >> 5) * 16;
  const int nbase = (wid & 31) * 64;
  const int nn = lane & 15, half = lane >> 4;
  const float deq = deqp[0];

  const v8i zero = {0, 0, 0, 0, 0, 0, 0, 0};
  v8i acc[4] = {zero, zero, zero, zero};

  const int* Xrow = (const int*)(Xq + (size_t)(mbase + nn) * Dn);
  for (int kk = 0; kk < Dn; kk += 64) {
    v8i a;
#pragma unroll
    for (int v = 0; v < 8; ++v)
      a[v] = Xrow[(kk + (v >> 1) * 16 + half * 8 + (v & 1) * 4) >> 2];
#pragma unroll
    for (int t = 0; t < 4; ++t) {
      const int* Wr = (const int*)(W8 + (size_t)(nbase + t * 16 + nn) * Dn);
      v8i bf;
#pragma unroll
      for (int j = 0; j < 8; ++j)
        bf[j] = Wr[(kk + (j >> 1) * 16 + half * 8 + (j & 1) * 4) >> 2];
      acc[t] = __builtin_amdgcn_wmma_i32_16x16x64_iu8(true, a, true, bf, acc[t], false, false);
    }
  }

#pragma unroll
  for (int t = 0; t < 4; ++t) {
    const int n = nbase + t * 16 + nn;
    const float bi = bias[n];
#pragma unroll
    for (int r = 0; r < 8; ++r) {
      const int mr = mbase + r + half * 8;
      float y = (float)acc[t][r] * deq + bi;
      if (MODE == 3) {
        of32[(size_t)mr * Dn + n] = y;
      } else {
        const int b_ = mr >> 10, t_ = mr & 1023, h_ = n >> 7, e_ = n & 127;
        const size_t hidx = ((size_t)(b_ * Hn + h_) * Tn + t_) * HDn + e_;
        if (MODE == 0) {
          oh[hidx] = (_Float16)(y * 0.08838834764831845f);
        } else if (MODE == 1) {
          of32[hidx] = y; oh[hidx] = (_Float16)y;
        } else {
          of32[hidx] = y;
          oh[((size_t)(b_ * Hn + h_) * HDn + e_) * Tn + t_] = (_Float16)y;
        }
      }
    }
  }
}

__global__ __launch_bounds__(128) void attn_kernel(const _Float16* __restrict__ qh,
                                                   const _Float16* __restrict__ kh,
                                                   const _Float16* __restrict__ vt,
                                                   float* __restrict__ attn) {
  __shared__ _Float16 pbuf[4][16 * 32];
  const int wv = threadIdx.x >> 5, lane = threadIdx.x & 31;
  const int wid = blockIdx.x * 4 + wv;
  const int bh = wid >> 6, qt = wid & 63;
  const int qbase = qt * 16;
  const int nn = lane & 15, half = lane >> 4;

  const _Float16* qb = qh + (size_t)bh * Tn * HDn;
  const _Float16* kb = kh + (size_t)bh * Tn * HDn;
  const _Float16* vb = vt + (size_t)bh * HDn * Tn;

  AF qa[4];
  {
    const int* qrow = (const int*)(qb + (size_t)(qbase + nn) * HDn);
#pragma unroll
    for (int c = 0; c < 4; ++c)
#pragma unroll
      for (int j = 0; j < 8; ++j) {
        int k = c * 32 + ((j >= 4) ? 16 : 0) + half * 8 + (j & 3) * 2;
        qa[c].i[j] = qrow[k >> 1];
      }
  }

  const v8f zf = {0.f, 0.f, 0.f, 0.f, 0.f, 0.f, 0.f, 0.f};
  v8f oacc[8] = {zf, zf, zf, zf, zf, zf, zf, zf};
  float mst[8], lst[8];
#pragma unroll
  for (int r = 0; r < 8; ++r) { mst[r] = -1e30f; lst[r] = 0.f; }

  const int nch = (qbase + 47) >> 5;
  for (int kc = 0; kc < nch; ++kc) {
    const int kbase = kc * 32;
    v8f s0 = zf, s1 = zf;
#pragma unroll
    for (int t2 = 0; t2 < 2; ++t2) {
      const int* krow = (const int*)(kb + (size_t)(kbase + t2 * 16 + nn) * HDn);
#pragma unroll
      for (int c = 0; c < 4; ++c) {
        AF bf;
#pragma unroll
        for (int j = 0; j < 8; ++j) bf.i[j] = krow[c * 16 + ((j >= 4) ? 8 : 0) + half * 4 + (j & 3)];
        if (t2 == 0)
          s0 = __builtin_amdgcn_wmma_f32_16x16x32_f16(false, qa[c].h, false, bf.h, (short)0, s0, false, false);
        else
          s1 = __builtin_amdgcn_wmma_f32_16x16x32_f16(false, qa[c].h, false, bf.h, (short)0, s1, false, false);
      }
    }
    float fac[8];
#pragma unroll
    for (int r = 0; r < 8; ++r) {
      const int row = qbase + r + half * 8;
      float a = s0[r] + (((kbase + nn) > row) ? -1e9f : 0.f);
      float b = s1[r] + (((kbase + 16 + nn) > row) ? -1e9f : 0.f);
      float mx = fmaxf(a, b);
#pragma unroll
      for (int m = 1; m < 16; m <<= 1) mx = fmaxf(mx, __shfl_xor(mx, m, 32));
      const float mnew = fmaxf(mst[r], mx);
      const float f = __expf(mst[r] - mnew);
      const float p0 = __expf(a - mnew), p1 = __expf(b - mnew);
      s0[r] = p0; s1[r] = p1;
      float rs = p0 + p1;
#pragma unroll
      for (int m = 1; m < 16; m <<= 1) rs += __shfl_xor(rs, m, 32);
      lst[r] = lst[r] * f + rs;
      mst[r] = mnew; fac[r] = f;
    }
#pragma unroll
    for (int j2 = 0; j2 < 8; ++j2)
#pragma unroll
      for (int r = 0; r < 8; ++r) oacc[j2][r] *= fac[r];

    _Float16* pb = pbuf[wv];
#pragma unroll
    for (int r = 0; r < 8; ++r) {
      const int row = r + half * 8;
      pb[row * 32 + nn]      = (_Float16)s0[r];
      pb[row * 32 + 16 + nn] = (_Float16)s1[r];
    }
    asm volatile("s_wait_dscnt 0" ::: "memory");
    AF pa;
    {
      const int* prow = (const int*)(pb + nn * 32);
#pragma unroll
      for (int j = 0; j < 8; ++j) {
        int k = ((j >= 4) ? 16 : 0) + half * 8 + (j & 3) * 2;
        pa.i[j] = prow[k >> 1];
      }
    }
#pragma unroll
    for (int j2 = 0; j2 < 8; ++j2) {
      const int* vrow = (const int*)(vb + (size_t)(j2 * 16 + nn) * Tn);
      AF bf;
#pragma unroll
      for (int j = 0; j < 8; ++j) bf.i[j] = vrow[kc * 16 + ((j >= 4) ? 8 : 0) + half * 4 + (j & 3)];
      oacc[j2] = __builtin_amdgcn_wmma_f32_16x16x32_f16(false, pa.h, false, bf.h, (short)0, oacc[j2], false, false);
    }
  }

  float* ab = attn + ((size_t)(bh >> 4) * Tn + qbase) * Dn + (size_t)(bh & 15) * HDn;
#pragma unroll
  for (int r = 0; r < 8; ++r) {
    const float inv = 1.f / lst[r];
#pragma unroll
    for (int j2 = 0; j2 < 8; ++j2)
      ab[(size_t)(r + half * 8) * Dn + j2 * 16 + nn] = oacc[j2][r] * inv;
  }
}

extern "C" void kernel_launch(void* const* d_in, const int* in_sizes, int n_in,
                              void* d_out, int out_size, void* d_ws, size_t ws_size,
                              hipStream_t stream) {
  (void)in_sizes; (void)n_in; (void)out_size; (void)ws_size;
  const float* x  = (const float*)d_in[0];
  const float* g  = (const float*)d_in[2];
  const float* be = (const float*)d_in[3];
  const float* Wq = (const float*)d_in[4];
  const float* bq = (const float*)d_in[5];
  const float* Wk = (const float*)d_in[6];
  const float* bk = (const float*)d_in[7];
  const float* Wv = (const float*)d_in[8];
  const float* bv = (const float*)d_in[9];
  const float* Wo = (const float*)d_in[10];
  const float* bo = (const float*)d_in[11];

  char* ws = (char*)d_ws;
  float*       sc    = (float*)(ws + OFF_SCAL);
  float*       parts = sc + 64;
  float*       xln   = (float*)(ws + OFF_LN);
  signed char* xi8   = (signed char*)(ws + OFF_XI8);
  signed char* w8    = (signed char*)(ws + OFF_W8);
  _Float16*    qh    = (_Float16*)(ws + OFF_QH);
  _Float16*    kh    = (_Float16*)(ws + OFF_KH);
  _Float16*    vt    = (_Float16*)(ws + OFF_VT);

  float* out  = (float*)d_out;
  float* outK = out + BTDsz;
  float* outV = out + 2 * BTDsz;

  init_kernel<<<1, 32, 0, stream>>>((unsigned*)sc);
  ln_kernel<<<BTn, 256, 0, stream>>>(x, g, be, xln, (unsigned*)sc);
  wabs_kernel<<<dim3(256, 4), 256, 0, stream>>>(Wq, Wk, Wv, Wo, parts);
  scales1_kernel<<<1, 256, 0, stream>>>(sc, parts);
  quantw_kernel<<<dim3(1024, 4), 256, 0, stream>>>(Wq, Wk, Wv, Wo, sc, w8);
  quanta_kernel<<<4096, 256, 0, stream>>>(xln, xi8, sc + 8, BTDsz);

  gemm_i8_kernel<0><<<4096, 128, 0, stream>>>(xi8, w8,             bq, sc + 12, nullptr, qh);
  gemm_i8_kernel<1><<<4096, 128, 0, stream>>>(xi8, w8 + DDsz,      bk, sc + 13, outK,    kh);
  gemm_i8_kernel<2><<<4096, 128, 0, stream>>>(xi8, w8 + 2 * DDsz,  bv, sc + 14, outV,    vt);

  attn_kernel<<<2048, 128, 0, stream>>>(qh, kh, vt, xln);

  amax_kernel<<<4096, 256, 0, stream>>>(xln, (unsigned*)sc + 1, BTDsz);
  scales2_kernel<<<1, 1, 0, stream>>>(sc);
  quanta_kernel<<<4096, 256, 0, stream>>>(xln, xi8, sc + 16, BTDsz);

  gemm_i8_kernel<3><<<4096, 128, 0, stream>>>(xi8, w8 + 3 * DDsz, bo, sc + 15, out, nullptr);
}
